// GNNModel_6098853560682
// MI455X (gfx1250) — matrix operand order repaired and checked by reading, not compiled, not run on hardware
//
#include <hip/hip_runtime.h>
#include <hip/hip_bf16.h>

typedef __attribute__((ext_vector_type(16))) __bf16 v16bf;
typedef __attribute__((ext_vector_type(4)))  __bf16 v4bf;
typedef __attribute__((ext_vector_type(8)))  float  v8f;

// ---------------- utility kernels ----------------

__global__ void gcn_zero_f32(float* __restrict__ p, int n) {
    int t = blockIdx.x * blockDim.x + threadIdx.x;
    if (t < n) p[t] = 0.0f;
}

// float4 -> 4x bf16 (8-byte store), n4 = n/4
__global__ void gcn_cvt_bf16x4(const float4* __restrict__ in, v4bf* __restrict__ out, int n4) {
    int t = blockIdx.x * blockDim.x + threadIdx.x;
    if (t >= n4) return;
    float4 f = in[t];
    v4bf o;
    o.x = (__bf16)f.x; o.y = (__bf16)f.y; o.z = (__bf16)f.z; o.w = (__bf16)f.w;
    out[t] = o;
}

// Repack a [K, Ncols] f32 weight matrix into per-lane WMMA B fragments (bf16):
// layout [ntileN][kb][lane][16 bf16], lane<16 -> col=lane, K=kb*32+{0..7,16..23} ; lane>=16 -> K=kb*32+{8..15,24..31}
__global__ void gcn_repack_w(const float* __restrict__ W, __bf16* __restrict__ Wp,
                             int Ncols, int nTilesN, int nKb) {
    int t = blockIdx.x * blockDim.x + threadIdx.x;
    int total = nTilesN * nKb * 32;
    if (t >= total) return;
    int ntile = t / (nKb * 32);
    int kb    = (t / 32) % nKb;
    int lane  = t % 32;
    int col   = ntile * 16 + (lane & 15);
    int k0    = kb * 32 + ((lane >> 4) * 8);
    __bf16* dst = Wp + (size_t)t * 16;
#pragma unroll
    for (int i = 0; i < 16; ++i)
        dst[i] = (__bf16)W[(size_t)(k0 + i + (i >= 8 ? 8 : 0)) * Ncols + col];
}

// ---------------- degree / norm ----------------

__global__ void gcn_degree(const int* __restrict__ dst, float* __restrict__ deg, int nEdges) {
    int e = blockIdx.x * blockDim.x + threadIdx.x;
    if (e < nEdges) atomicAdd(deg + dst[e], 1.0f);
}

__global__ void gcn_dinv(float* __restrict__ deg, int nNodes) {
    int i = blockIdx.x * blockDim.x + threadIdx.x;
    if (i < nNodes) deg[i] = rsqrtf(deg[i] + 1.0f);   // in-place: deg -> dinv
}

// ---------------- WMMA GEMM: C[M,N] = A[M,K](bf16) * Bp(packed bf16), f32 accum ----------------
// NT = number of 16-wide N tiles (= waves per block), KB = number of 32-deep K steps.
template <int NT, int KB>
__global__ __launch_bounds__(NT * 32) void gcn_gemm_wmma(const __bf16* __restrict__ A,
                                                         const __bf16* __restrict__ Bp,
                                                         float* __restrict__ C) {
    constexpr int K = KB * 32;
    constexpr int N = NT * 16;
    constexpr int NV = 16 * K * 2 / 16;        // uint4 count for a 16-row A tile
    __shared__ uint4 lds4[NV];

    const int m0 = blockIdx.x * 16;

    // Async DMA of the contiguous 16xK bf16 tile into LDS (ASYNCcnt path).
    // Generic LDS pointer's low 32 bits are the LDS byte offset; global side is
    // GVS addressing: 64-bit SGPR base + 32-bit per-lane byte offset.
    const unsigned gbase = (unsigned)((size_t)m0 * K * 2);
    for (int i = threadIdx.x; i < NV; i += NT * 32) {
        unsigned ldsAddr = (unsigned)(uintptr_t)(lds4 + i);
        unsigned goff    = gbase + (unsigned)i * 16u;
        asm volatile("global_load_async_to_lds_b128 %0, %1, %2"
                     :: "v"(ldsAddr), "v"(goff), "s"(A)
                     : "memory");
    }
    asm volatile("s_wait_asynccnt 0x0" ::: "memory");
    __syncthreads();
    const __bf16* lds = reinterpret_cast<const __bf16*>(lds4);

    const int lane = threadIdx.x & 31;
    const int wave = threadIdx.x >> 5;         // which N tile
    const int arow = lane & 15;
    const int asel = (lane >> 4) * 8;          // K sub-offset per ISA A-fragment layout

    v8f acc = {};
#pragma unroll
    for (int kb = 0; kb < KB; ++kb) {
        const int koff = kb * 32 + asel;
        union { v16bf v; uint4 u[2]; } a, b;
        a.u[0] = *reinterpret_cast<const uint4*>(lds + arow * K + koff);        // K+0..7
        a.u[1] = *reinterpret_cast<const uint4*>(lds + arow * K + koff + 16);   // K+16..23
        const uint4* bp = reinterpret_cast<const uint4*>(
            Bp + (size_t)((wave * KB + kb) * 32 + lane) * 16);
        b.u[0] = bp[0];
        b.u[1] = bp[1];
        acc = __builtin_amdgcn_wmma_f32_16x16x32_bf16(false, a.v, false, b.v,
                                                      (short)0, acc, false, false);
    }

    const int col   = wave * 16 + (lane & 15);
    const int rbase = (lane >> 4) * 8;         // C fragment: VGPR r -> row r (+8 for hi lanes)
#pragma unroll
    for (int r = 0; r < 8; ++r)
        C[(size_t)(m0 + rbase + r) * N + col] = acc[r];
}

// ---------------- edge scatter: out[dst] += dinv[src]*dinv[dst] * msg[src] ----------------
// C4 = channels/4 threads per edge, each thread handles a float4 chunk.
template <int C4>
__global__ void gcn_aggregate(const int* __restrict__ src, const int* __restrict__ dst,
                              const float* __restrict__ dinv, const float* __restrict__ msg,
                              float* __restrict__ out, int nEdges) {
    constexpr int C = C4 * 4;
    int t = blockIdx.x * blockDim.x + threadIdx.x;
    int e = t / C4;
    if (e >= nEdges) return;
    int c4 = t % C4;
    int s = src[e], d = dst[e];
    float nrm = dinv[s] * dinv[d];
    const float4 m = *reinterpret_cast<const float4*>(msg + (size_t)s * C + c4 * 4);
    float* o = out + (size_t)d * C + c4 * 4;
    atomicAdd(o + 0, nrm * m.x);
    atomicAdd(o + 1, nrm * m.y);
    atomicAdd(o + 2, nrm * m.z);
    atomicAdd(o + 3, nrm * m.w);
}

// ---------------- post ops ----------------

// h = relu(agg + dinv^2 * xw + b)  -> bf16 for the next GEMM
__global__ void gcn_post1(const float* __restrict__ agg, const float* __restrict__ xw,
                          const float* __restrict__ dinv, const float* __restrict__ b1,
                          __bf16* __restrict__ hbf, int nNodes) {
    int t = blockIdx.x * blockDim.x + threadIdx.x;
    if (t >= nNodes * 64) return;
    int i = t >> 6, c = t & 63;
    float di = dinv[i];
    float h = agg[t] + di * di * xw[t] + b1[c];
    hbf[t] = (__bf16)fmaxf(h, 0.0f);
}

// out = out(agg) + dinv^2 * hw2 + b2
__global__ void gcn_post2(float* __restrict__ out, const float* __restrict__ hw,
                          const float* __restrict__ dinv, const float* __restrict__ b2,
                          int nNodes) {
    int t = blockIdx.x * blockDim.x + threadIdx.x;
    if (t >= nNodes * 32) return;
    int i = t >> 5, c = t & 31;
    float di = dinv[i];
    out[t] = out[t] + di * di * hw[t] + b2[c];
}

// ---------------- launcher ----------------

extern "C" void kernel_launch(void* const* d_in, const int* in_sizes, int n_in,
                              void* d_out, int out_size, void* d_ws, size_t ws_size,
                              hipStream_t stream) {
    const float* x  = (const float*)d_in[0];
    const int*   ei = (const int*)d_in[1];
    const float* W1 = (const float*)d_in[2];
    const float* b1 = (const float*)d_in[3];
    const float* W2 = (const float*)d_in[4];
    const float* b2 = (const float*)d_in[5];
    float* out = (float*)d_out;

    const int nNodes = in_sizes[0] / 128;
    const int nEdges = in_sizes[1] / 2;
    const int* esrc = ei;
    const int* edst = ei + nEdges;

    // workspace carve-up (256B aligned regions)
    char* ws = (char*)d_ws;
    size_t off = 0;
    auto alloc = [&](size_t bytes) -> void* {
        void* p = ws + off;
        off = (off + bytes + 255) & ~(size_t)255;
        return p;
    };
    float*  dinv = (float*)alloc((size_t)nNodes * 4);
    __bf16* xbf  = (__bf16*)alloc((size_t)nNodes * 128 * 2);  // later aliased as agg1 (same size)
    float*  agg1 = (float*)xbf;
    float*  xw1  = (float*)alloc((size_t)nNodes * 64 * 4);    // later aliased as hw2 (smaller)
    float*  hw2  = xw1;
    __bf16* hbf  = (__bf16*)alloc((size_t)nNodes * 64 * 2);
    __bf16* w1p  = (__bf16*)alloc((size_t)4 * 4 * 32 * 16 * 2);
    __bf16* w2p  = (__bf16*)alloc((size_t)2 * 2 * 32 * 16 * 2);

    const int B = 256;
    auto nb = [&](long long n) { return (int)((n + B - 1) / B); };

    // degrees -> dinv
    gcn_zero_f32<<<nb(nNodes), B, 0, stream>>>(dinv, nNodes);
    gcn_degree<<<nb(nEdges), B, 0, stream>>>(edst, dinv, nEdges);
    gcn_dinv<<<nb(nNodes), B, 0, stream>>>(dinv, nNodes);

    // precision conversion + weight repack
    gcn_cvt_bf16x4<<<nb((long long)nNodes * 32), B, 0, stream>>>(
        (const float4*)x, (v4bf*)xbf, nNodes * 32);
    gcn_repack_w<<<nb(4 * 4 * 32), B, 0, stream>>>(W1, w1p, 64, 4, 4);
    gcn_repack_w<<<nb(2 * 2 * 32), B, 0, stream>>>(W2, w2p, 32, 2, 2);

    // layer 1: GEMM (WMMA) -> scatter -> self-loop+bias+relu -> bf16
    gcn_gemm_wmma<4, 4><<<nNodes / 16, 128, 0, stream>>>(xbf, w1p, xw1);
    gcn_zero_f32<<<nb((long long)nNodes * 64), B, 0, stream>>>(agg1, nNodes * 64);  // xbf dead now
    gcn_aggregate<16><<<nb((long long)nEdges * 16), B, 0, stream>>>(esrc, edst, dinv, xw1, agg1, nEdges);
    gcn_post1<<<nb((long long)nNodes * 64), B, 0, stream>>>(agg1, xw1, dinv, b1, hbf, nNodes);

    // layer 2: GEMM (WMMA) -> scatter into d_out -> self-loop+bias
    gcn_gemm_wmma<2, 2><<<nNodes / 16, 64, 0, stream>>>(hbf, w2p, hw2);  // xw1 dead now
    gcn_zero_f32<<<nb((long long)nNodes * 32), B, 0, stream>>>(out, nNodes * 32);
    gcn_aggregate<8><<<nb((long long)nEdges * 8), B, 0, stream>>>(esrc, edst, dinv, hw2, out, nEdges);
    gcn_post2<<<nb((long long)nNodes * 32), B, 0, stream>>>(out, hw2, dinv, b2, nNodes);
}
